// CausalSelfAttention_12592844112122
// MI455X (gfx1250) — matrix operand order repaired and checked by reading, not compiled, not run on hardware
//
#include <hip/hip_runtime.h>

// ---------- types ----------
typedef __attribute__((ext_vector_type(16))) __bf16 v16bf;
typedef __attribute__((ext_vector_type(8)))  __bf16 v8bf;
typedef __attribute__((ext_vector_type(8)))  float  v8f;

union un16 { v16bf v; v8bf h[2]; };

#define WMMA_BF16(a, b, c) \
  __builtin_amdgcn_wmma_f32_16x16x32_bf16(false, (a), false, (b), (short)0, (c), false, false)

// Problem constants
#define BB    2
#define TT    2048
#define CC    1024
#define HH    16
#define DD    64
#define MM    (BB * TT)        // 4096
#define LDSS  40               // 32 + 8 pad (bf16 elems)

// ---------- CDNA5 async global->LDS copy (ASYNCcnt-tracked, bypasses VGPRs) ----------
// Generic pointers to LDS carry the wave-relative LDS byte offset in addr[31:0]
// (ISA 10.2 aperture mapping), which is exactly what VDST of the async op wants.
__device__ __forceinline__ void async_copy_b128(void* lds_ptr, const void* gptr) {
  unsigned l = (unsigned)(size_t)lds_ptr;
  asm volatile("global_load_async_to_lds_b128 %0, %1, off"
               :: "v"(l), "v"(gptr) : "memory");
}
__device__ __forceinline__ void wait_async0() {
  asm volatile("s_wait_asynccnt 0x0" ::: "memory");
}

// ---------- small reductions over the 16-lane half that owns a C-fragment row ----------
__device__ __forceinline__ float rowmax16(float v) {
  v = fmaxf(v, __shfl_xor(v, 1, 32));
  v = fmaxf(v, __shfl_xor(v, 2, 32));
  v = fmaxf(v, __shfl_xor(v, 4, 32));
  v = fmaxf(v, __shfl_xor(v, 8, 32));
  return v;
}
__device__ __forceinline__ float rowsum16(float v) {
  v += __shfl_xor(v, 1, 32);
  v += __shfl_xor(v, 2, 32);
  v += __shfl_xor(v, 4, 32);
  v += __shfl_xor(v, 8, 32);
  return v;
}

// ---------- conversion kernels ----------
__global__ void cvt_bf16_kernel(const float* __restrict__ in, __bf16* __restrict__ out, int n) {
  int i = blockIdx.x * blockDim.x + threadIdx.x;
  if (i < n) out[i] = (__bf16)in[i];
}

// in [K][N] fp32 -> out [N][K] bf16
__global__ void transpose_bf16_kernel(const float* __restrict__ in, __bf16* __restrict__ out,
                                      int K, int N) {
  int i = blockIdx.x * blockDim.x + threadIdx.x;
  if (i < K * N) {
    int k = i / N, n = i % N;
    out[(size_t)n * K + k] = (__bf16)in[i];
  }
}

// ---------- QKV GEMM: [4096,1024] x [1024,3072] + bias -> scattered Q/K/Vt bf16 ----------
__global__ __launch_bounds__(256) void gemm_qkv_kernel(
    const __bf16* __restrict__ A,    // xb [M][1024]
    const __bf16* __restrict__ Bt,   // wqkvT [3072][1024]
    const float*  __restrict__ bias, // [3072]
    __bf16* __restrict__ Qb, __bf16* __restrict__ Kb, __bf16* __restrict__ Vt) {
  __shared__ __bf16 sA[2][128 * LDSS];
  __shared__ __bf16 sB[2][128 * LDSS];
  const int K = CC;
  const int tile_m = blockIdx.x * 128;
  const int tile_n = blockIdx.y * 128;
  const int tid  = threadIdx.x;
  const int lane = tid & 31, wid = tid >> 5;
  const int wm = (wid & 1) * 64, wn = (wid >> 1) * 32;
  const int hi  = lane >> 4;
  const int l15 = lane & 15;

  // this thread's two (row, col8) chunks for cooperative tile loads
  const int row0 = tid >> 2,           col0 = (tid & 3) * 8;
  const int row1 = (tid + 256) >> 2,   col1 = (tid & 3) * 8;   // (c&3) identical for +256

  v8f acc[4][2];
#pragma unroll
  for (int mi = 0; mi < 4; ++mi)
#pragma unroll
    for (int ni = 0; ni < 2; ++ni)
#pragma unroll
      for (int r = 0; r < 8; ++r) acc[mi][ni][r] = 0.f;

  // prologue: async-load tile k0=0 into buffer 0
  async_copy_b128(&sA[0][row0 * LDSS + col0], &A[(size_t)(tile_m + row0) * K + col0]);
  async_copy_b128(&sA[0][row1 * LDSS + col1], &A[(size_t)(tile_m + row1) * K + col1]);
  async_copy_b128(&sB[0][row0 * LDSS + col0], &Bt[(size_t)(tile_n + row0) * K + col0]);
  async_copy_b128(&sB[0][row1 * LDSS + col1], &Bt[(size_t)(tile_n + row1) * K + col1]);
  wait_async0();
  __syncthreads();

  int buf = 0;
  for (int k0 = 0; k0 < K; k0 += 32) {
    // prefetch next k-tile into the other buffer (async, overlaps WMMA below)
    if (k0 + 32 < K) {
      int kn = k0 + 32;
      async_copy_b128(&sA[buf ^ 1][row0 * LDSS + col0], &A[(size_t)(tile_m + row0) * K + kn + col0]);
      async_copy_b128(&sA[buf ^ 1][row1 * LDSS + col1], &A[(size_t)(tile_m + row1) * K + kn + col1]);
      async_copy_b128(&sB[buf ^ 1][row0 * LDSS + col0], &Bt[(size_t)(tile_n + row0) * K + kn + col0]);
      async_copy_b128(&sB[buf ^ 1][row1 * LDSS + col1], &Bt[(size_t)(tile_n + row1) * K + kn + col1]);
    }
    v16bf afr[4], bfr[2];
#pragma unroll
    for (int mi = 0; mi < 4; ++mi) {
      const __bf16* p = &sA[buf][(wm + mi * 16 + l15) * LDSS];
      un16 u; u.h[0] = *(const v8bf*)(p + 8 * hi); u.h[1] = *(const v8bf*)(p + 16 + 8 * hi);
      afr[mi] = u.v;
    }
#pragma unroll
    for (int ni = 0; ni < 2; ++ni) {
      const __bf16* p = &sB[buf][(wn + ni * 16 + l15) * LDSS + 8 * hi];
      un16 u; u.h[0] = *(const v8bf*)p; u.h[1] = *(const v8bf*)(p + 16);
      bfr[ni] = u.v;
    }
#pragma unroll
    for (int mi = 0; mi < 4; ++mi)
#pragma unroll
      for (int ni = 0; ni < 2; ++ni)
        acc[mi][ni] = WMMA_BF16(afr[mi], bfr[ni], acc[mi][ni]);
    wait_async0();
    __syncthreads();
    buf ^= 1;
  }

#pragma unroll
  for (int mi = 0; mi < 4; ++mi)
#pragma unroll
    for (int ni = 0; ni < 2; ++ni)
#pragma unroll
      for (int r = 0; r < 8; ++r) {
        int gm = tile_m + wm + mi * 16 + r + 8 * hi;
        int gn = tile_n + wn + ni * 16 + l15;
        float v = acc[mi][ni][r] + bias[gn];
        int bb = gm >> 11, tt = gm & (TT - 1);
        int which = gn >> 10;
        int n = gn & (CC - 1);
        int h = n >> 6, d = n & 63;
        size_t bh = (size_t)(bb * HH + h);
        __bf16 bv = (__bf16)v;
        if (which == 0)      Qb[(bh * TT + tt) * DD + d] = bv;
        else if (which == 1) Kb[(bh * TT + tt) * DD + d] = bv;
        else                 Vt[(bh * DD + d) * TT + tt] = bv;
      }
}

// ---------- flash attention: per (b,h), 64 q-rows per block, 4 waves ----------
__global__ __launch_bounds__(128) void attn_kernel(
    const __bf16* __restrict__ Qb, const __bf16* __restrict__ Kb,
    const __bf16* __restrict__ Vt, __bf16* __restrict__ Yb) {
  const int bh   = blockIdx.x;     // 0..31
  const int qblk = blockIdx.y;     // 0..31
  const int lane = threadIdx.x & 31, w = threadIdx.x >> 5;
  const int hi = lane >> 4, l15 = lane & 15;
  const int q0 = qblk * 64 + w * 16;

  const __bf16* Qp = Qb + (size_t)bh * TT * DD;
  const __bf16* Kp = Kb + (size_t)bh * TT * DD;
  const __bf16* Vp = Vt + (size_t)bh * DD * TT;

  __shared__ __bf16 sP[4][16 * LDSS];   // per-wave P staging

  // Q row-block held in registers: two 16x32 A-fragments
  v16bf qf[2];
#pragma unroll
  for (int kf = 0; kf < 2; ++kf) {
    const __bf16* p = Qp + (size_t)(q0 + l15) * DD + kf * 32 + 8 * hi;
    un16 u; u.h[0] = *(const v8bf*)p; u.h[1] = *(const v8bf*)(p + 16);
    qf[kf] = u.v;
  }

  v8f o[4];
#pragma unroll
  for (int j = 0; j < 4; ++j)
#pragma unroll
    for (int r = 0; r < 8; ++r) o[j][r] = 0.f;
  float m_r[8], l_r[8];
#pragma unroll
  for (int r = 0; r < 8; ++r) { m_r[r] = -1e30f; l_r[r] = 0.f; }

  const int nkt = (q0 + 15) / 32 + 1;   // causal extent for this wave
  for (int kt = 0; kt < nkt; ++kt) {
    const int kbase = kt * 32;
    // hint-prefetch next tile of K and V while we compute this one
    if (kt + 1 < nkt) {
      __builtin_prefetch(Kp + (size_t)(kbase + 32 + l15) * DD, 0, 0);
      __builtin_prefetch(Vp + (size_t)l15 * TT + kbase + 32, 0, 0);
    }
    // K fragments: [key-chunk][d-chunk]
    v16bf kfr[2][2];
#pragma unroll
    for (int kc = 0; kc < 2; ++kc)
#pragma unroll
      for (int df = 0; df < 2; ++df) {
        const __bf16* p = Kp + (size_t)(kbase + kc * 16 + l15) * DD + df * 32 + 8 * hi;
        un16 u; u.h[0] = *(const v8bf*)p; u.h[1] = *(const v8bf*)(p + 16);
        kfr[kc][df] = u.v;
      }
    v8f s[2];
#pragma unroll
    for (int kc = 0; kc < 2; ++kc) {
#pragma unroll
      for (int r = 0; r < 8; ++r) s[kc][r] = 0.f;
      s[kc] = WMMA_BF16(qf[0], kfr[kc][0], s[kc]);
      s[kc] = WMMA_BF16(qf[1], kfr[kc][1], s[kc]);
    }
    // scale, causal mask, online softmax, stage P
#pragma unroll
    for (int r = 0; r < 8; ++r) {
      int row = q0 + r + 8 * hi;
      float s0 = s[0][r] * 0.125f;
      float s1 = s[1][r] * 0.125f;
      if (kbase + l15 > row)      s0 = -1e30f;
      if (kbase + 16 + l15 > row) s1 = -1e30f;
      float mt = rowmax16(fmaxf(s0, s1));
      float mn = fmaxf(m_r[r], mt);
      float corr = __expf(m_r[r] - mn);
      float p0 = __expf(s0 - mn);
      float p1 = __expf(s1 - mn);
      float ps = rowsum16(p0 + p1);
      l_r[r] = l_r[r] * corr + ps;
      m_r[r] = mn;
#pragma unroll
      for (int j = 0; j < 4; ++j) o[j][r] *= corr;
      __bf16* pr = &sP[w][(r + 8 * hi) * LDSS];
      pr[l15]      = (__bf16)p0;
      pr[16 + l15] = (__bf16)p1;
    }
    // reload P as a 16x32 A-fragment (per-wave region; DS in-order within wave)
    const __bf16* pp = &sP[w][l15 * LDSS + 8 * hi];
    un16 up; up.h[0] = *(const v8bf*)pp; up.h[1] = *(const v8bf*)(pp + 16);
    v16bf pf = up.v;
    // P @ V: four 32x16 B-fragments from Vt (t contiguous)
#pragma unroll
    for (int j = 0; j < 4; ++j) {
      const __bf16* vp = Vp + (size_t)(j * 16 + l15) * TT + kbase + 8 * hi;
      un16 uv; uv.h[0] = *(const v8bf*)vp; uv.h[1] = *(const v8bf*)(vp + 16);
      o[j] = WMMA_BF16(pf, uv.v, o[j]);
    }
  }

  const int b = bh >> 4, h = bh & (HH - 1);
#pragma unroll
  for (int j = 0; j < 4; ++j)
#pragma unroll
    for (int r = 0; r < 8; ++r) {
      int row = q0 + r + 8 * hi;
      float val = o[j][r] / l_r[r];
      size_t gm = (size_t)b * TT + row;
      Yb[gm * CC + h * DD + j * 16 + l15] = (__bf16)val;
    }
}

// ---------- output projection: Yb[4096,1024] x wprojT -> fp32 out ----------
__global__ __launch_bounds__(256) void gemm_proj_kernel(
    const __bf16* __restrict__ A,    // Yb [M][1024]
    const __bf16* __restrict__ Bt,   // wprojT [1024][1024]
    const float*  __restrict__ bias, // [1024]
    float* __restrict__ out) {       // [M][1024]
  __shared__ __bf16 sA[2][128 * LDSS];
  __shared__ __bf16 sB[2][128 * LDSS];
  const int K = CC;
  const int tile_m = blockIdx.x * 128;
  const int tile_n = blockIdx.y * 128;
  const int tid  = threadIdx.x;
  const int lane = tid & 31, wid = tid >> 5;
  const int wm = (wid & 1) * 64, wn = (wid >> 1) * 32;
  const int hi  = lane >> 4;
  const int l15 = lane & 15;
  const int row0 = tid >> 2,         col0 = (tid & 3) * 8;
  const int row1 = (tid + 256) >> 2, col1 = (tid & 3) * 8;

  v8f acc[4][2];
#pragma unroll
  for (int mi = 0; mi < 4; ++mi)
#pragma unroll
    for (int ni = 0; ni < 2; ++ni)
#pragma unroll
      for (int r = 0; r < 8; ++r) acc[mi][ni][r] = 0.f;

  async_copy_b128(&sA[0][row0 * LDSS + col0], &A[(size_t)(tile_m + row0) * K + col0]);
  async_copy_b128(&sA[0][row1 * LDSS + col1], &A[(size_t)(tile_m + row1) * K + col1]);
  async_copy_b128(&sB[0][row0 * LDSS + col0], &Bt[(size_t)(tile_n + row0) * K + col0]);
  async_copy_b128(&sB[0][row1 * LDSS + col1], &Bt[(size_t)(tile_n + row1) * K + col1]);
  wait_async0();
  __syncthreads();

  int buf = 0;
  for (int k0 = 0; k0 < K; k0 += 32) {
    if (k0 + 32 < K) {
      int kn = k0 + 32;
      async_copy_b128(&sA[buf ^ 1][row0 * LDSS + col0], &A[(size_t)(tile_m + row0) * K + kn + col0]);
      async_copy_b128(&sA[buf ^ 1][row1 * LDSS + col1], &A[(size_t)(tile_m + row1) * K + kn + col1]);
      async_copy_b128(&sB[buf ^ 1][row0 * LDSS + col0], &Bt[(size_t)(tile_n + row0) * K + kn + col0]);
      async_copy_b128(&sB[buf ^ 1][row1 * LDSS + col1], &Bt[(size_t)(tile_n + row1) * K + kn + col1]);
    }
    v16bf afr[4], bfr[2];
#pragma unroll
    for (int mi = 0; mi < 4; ++mi) {
      const __bf16* p = &sA[buf][(wm + mi * 16 + l15) * LDSS];
      un16 u; u.h[0] = *(const v8bf*)(p + 8 * hi); u.h[1] = *(const v8bf*)(p + 16 + 8 * hi);
      afr[mi] = u.v;
    }
#pragma unroll
    for (int ni = 0; ni < 2; ++ni) {
      const __bf16* p = &sB[buf][(wn + ni * 16 + l15) * LDSS + 8 * hi];
      un16 u; u.h[0] = *(const v8bf*)p; u.h[1] = *(const v8bf*)(p + 16);
      bfr[ni] = u.v;
    }
#pragma unroll
    for (int mi = 0; mi < 4; ++mi)
#pragma unroll
      for (int ni = 0; ni < 2; ++ni)
        acc[mi][ni] = WMMA_BF16(afr[mi], bfr[ni], acc[mi][ni]);
    wait_async0();
    __syncthreads();
    buf ^= 1;
  }

#pragma unroll
  for (int mi = 0; mi < 4; ++mi)
#pragma unroll
    for (int ni = 0; ni < 2; ++ni)
#pragma unroll
      for (int r = 0; r < 8; ++r) {
        int gm = tile_m + wm + mi * 16 + r + 8 * hi;
        int gn = tile_n + wn + ni * 16 + l15;
        out[(size_t)gm * CC + gn] = acc[mi][ni][r] + bias[gn];
      }
}

// ---------- host launcher ----------
extern "C" void kernel_launch(void* const* d_in, const int* in_sizes, int n_in,
                              void* d_out, int out_size, void* d_ws, size_t ws_size,
                              hipStream_t stream) {
  (void)in_sizes; (void)n_in; (void)out_size; (void)ws_size;
  const float* x      = (const float*)d_in[0];
  const float* w_qkv  = (const float*)d_in[1];
  const float* b_qkv  = (const float*)d_in[2];
  const float* w_proj = (const float*)d_in[3];
  const float* b_proj = (const float*)d_in[4];
  float* out = (float*)d_out;

  char* ws = (char*)d_ws;
  __bf16* xb     = (__bf16*)(ws + 0);              //  8,388,608 B
  __bf16* wqkvT  = (__bf16*)(ws + 8388608);        //  6,291,456 B
  __bf16* wprojT = (__bf16*)(ws + 14680064);       //  2,097,152 B
  __bf16* Qb     = (__bf16*)(ws + 16777216);       //  8,388,608 B
  __bf16* Kb     = (__bf16*)(ws + 25165824);       //  8,388,608 B
  __bf16* Vt     = (__bf16*)(ws + 33554432);       //  8,388,608 B
  __bf16* Yb     = (__bf16*)(ws + 41943040);       //  8,388,608 B (total 48 MB)

  {
    int n = MM * CC;
    cvt_bf16_kernel<<<(n + 255) / 256, 256, 0, stream>>>(x, xb, n);
  }
  {
    int n = CC * 3 * CC;
    transpose_bf16_kernel<<<(n + 255) / 256, 256, 0, stream>>>(w_qkv, wqkvT, CC, 3 * CC);
  }
  {
    int n = CC * CC;
    transpose_bf16_kernel<<<(n + 255) / 256, 256, 0, stream>>>(w_proj, wprojT, CC, CC);
  }
  gemm_qkv_kernel<<<dim3(MM / 128, (3 * CC) / 128), 256, 0, stream>>>(xb, wqkvT, b_qkv, Qb, Kb, Vt);
  attn_kernel<<<dim3(BB * HH, TT / 64), 128, 0, stream>>>(Qb, Kb, Vt, Yb);
  gemm_proj_kernel<<<dim3(MM / 128, CC / 128), 256, 0, stream>>>(Yb, wprojT, b_proj, out);
}
